// SelfAttention_37349035606256
// MI455X (gfx1250) — compile-verified, matrix operand order checked by reading, not run on hardware
//
#include <hip/hip_runtime.h>

// MI455X / gfx1250, wave32.  WMMA 16x16x32 bf16 (f32 accum) + TDM double-
// buffered staging (tensor_load_to_lds, TENSORcnt).

typedef __attribute__((ext_vector_type(16))) __bf16 v16bf;
typedef __attribute__((ext_vector_type(8)))  float  v8f;
typedef unsigned int u32x4 __attribute__((ext_vector_type(4)));
typedef int          i32x8 __attribute__((ext_vector_type(8)));
typedef int          i32x4 __attribute__((ext_vector_type(4)));

#define NBATCH 8
#define SEQ    1024
#define EMB    1024
#define HEADS  16
#define HDIM   64
#define CHUNK  32

// bf16 fragment from a row-major bf16 LDS row (A/B layout for 16x16x32):
// lanes 0-15 take K={0..7,16..23}, lanes 16-31 take K={8..15,24..31}.
#define LOAD_FRAG(dst, rowptr, colbase)                     \
  do {                                                      \
    _Pragma("unroll") for (int _j = 0; _j < 8; ++_j) {      \
      (dst)[_j]     = (rowptr)[(colbase) + kb + _j];        \
      (dst)[_j + 8] = (rowptr)[(colbase) + kb + 16 + _j];   \
    }                                                       \
  } while (0)

// Same fragment layout, but source row is fp32 in LDS (TDM-staged tile);
// convert on read (v_cvt_pk_bf16_f32 pairs, co-executes with WMMA).
#define LOAD_FRAG_CVT(dst, rowptr, colbase)                          \
  do {                                                               \
    _Pragma("unroll") for (int _j = 0; _j < 8; ++_j) {               \
      (dst)[_j]     = (__bf16)(rowptr)[(colbase) + kb + _j];         \
      (dst)[_j + 8] = (__bf16)(rowptr)[(colbase) + kb + 16 + _j];    \
    }                                                                \
  } while (0)

#define WMMA_BF16(a, b, c) \
  __builtin_amdgcn_wmma_f32_16x16x32_bf16(false, (a), false, (b), (short)0, (c), false, false)

// ---------------------------------------------------------------------------
// TDM: issue a 2D tile load (fp32/int32 elements) global -> LDS.
// D# per CDNA5 ISA ch.8: group0 = {count/type/lds_addr/global_addr},
// group1 = {data_size, tensor dims, tile dims, strides}.  Tensor dims are set
// large (tiles are always in-bounds).  2D: groups 2/3 (and the trailing
// extension group of the 6-arg builtin) are zero.
// ---------------------------------------------------------------------------
static __device__ __forceinline__ void tdm_load_2d(void* lds_dst, const void* gsrc,
                                                   unsigned tile_w, unsigned tile_h,
                                                   unsigned stride_elems) {
  // LDS aperture keeps the byte offset in addr[31:0] (ISA 10.2), so the low
  // 32 bits of the generic pointer are the LDS address for the descriptor.
  unsigned lds_byte = (unsigned)(size_t)lds_dst;
  unsigned long long ga = (unsigned long long)(size_t)gsrc;

  u32x4 g0;
  g0.x = 1u;                                                  // count=1 (valid), no gather
  g0.y = lds_byte;                                            // lds_addr
  g0.z = (unsigned)ga;                                        // global_addr[31:0]
  g0.w = (unsigned)((ga >> 32) & 0x01FFFFFFull) | 0x80000000u; // addr[56:32] | type=2

  const unsigned td0 = 0x100000u, td1 = 0x100000u;            // huge tensor dims (no OOB)
  i32x8 g1;
  g1[0] = (int)(2u << 16);                                    // wg_mask=0, data_size=4B
  g1[1] = (int)((td0 & 0xFFFFu) << 16);                       // tensor_dim0[15:0]
  g1[2] = (int)(((td0 >> 16) & 0xFFFFu) | ((td1 & 0xFFFFu) << 16));
  g1[3] = (int)(((td1 >> 16) & 0xFFFFu) | ((tile_w & 0xFFFFu) << 16)); // tile_dim0
  g1[4] = (int)(tile_h & 0xFFFFu);                            // tile_dim1; tile_dim2=0
  g1[5] = (int)stride_elems;                                  // tensor_dim0_stride[31:0]
  g1[6] = 0;                                                  // stride0 hi / stride1 lo
  g1[7] = 0;

  i32x4 z4 = {0, 0, 0, 0};
  i32x8 z8 = {0, 0, 0, 0, 0, 0, 0, 0};
  __builtin_amdgcn_tensor_load_to_lds(g0, g1, z4, z4, z8, 0);
}

// ---------------------------------------------------------------------------
// Kernel 1: fused masked attention.  grid = (N*H, SEQ/64), block = 128.
// Wave w owns q rows [blockIdx.y*64 + w*16, +16); TDM double-buffers the
// K chunk (32x64 f32), V chunk (32x64 f32) and mask tile (64x32 i32) in LDS.
// ---------------------------------------------------------------------------
__global__ __launch_bounds__(128)
void attn_fused_kernel(const float* __restrict__ Vp, const float* __restrict__ Kp,
                       const float* __restrict__ Qp, const int* __restrict__ Msk,
                       float* __restrict__ X) {
  __shared__ float  KlB[2][CHUNK][HDIM];   // 16 KB
  __shared__ float  VlB[2][CHUNK][HDIM];   // 16 KB
  __shared__ int    MlB[2][64][CHUNK];     // 16 KB
  __shared__ __bf16 Pl[4][16][CHUNK];      //  4 KB (per-wave P tiles)

  const int tid  = threadIdx.x;
  const int wave = tid >> 5;
  const int lane = tid & 31;
  const int lq   = lane & 15;
  const int half = lane >> 4;
  const int kb   = half * 8;

  const int n     = blockIdx.x >> 4;
  const int h     = blockIdx.x & 15;
  const int qg0   = blockIdx.y * 64;       // workgroup q base
  const int qbase = qg0 + wave * 16;       // wave q base

  const size_t head_off = (size_t)h * HDIM;
  const float  inv_scale = 0.03125f;       // 1/sqrt(EMB) = 1/32 (reference quirk)

  // Q fragments (registers, whole kernel): two head-dim k-chunks.
  v16bf aQ0, aQ1;
  {
    const float* qrow = Qp + ((size_t)n * SEQ + qbase + lq) * EMB + head_off;
#pragma unroll
    for (int j = 0; j < 8; ++j) {
      aQ0[j]     = (__bf16)qrow[kb + j];
      aQ0[j + 8] = (__bf16)qrow[kb + 16 + j];
      aQ1[j]     = (__bf16)qrow[32 + kb + j];
      aQ1[j + 8] = (__bf16)qrow[32 + kb + 16 + j];
    }
  }

  v8f acc[4] = {{}, {}, {}, {}};
  float mrow[8], lrow[8];
#pragma unroll
  for (int r = 0; r < 8; ++r) { mrow[r] = -3.0e38f; lrow[r] = 0.0f; }

  const float* kbase_p = Kp + ((size_t)n * SEQ) * EMB + head_off;
  const float* vbase_p = Vp + ((size_t)n * SEQ) * EMB + head_off;
  const int*   mbase_p = Msk + (size_t)n * SEQ * SEQ + (size_t)qg0 * SEQ;

  // Prologue: TDM chunk 0 into buffer 0 (wave 0 issues; TENSORcnt tracks).
  if (wave == 0) {
    tdm_load_2d(&KlB[0][0][0], kbase_p, HDIM, CHUNK, EMB);
    tdm_load_2d(&VlB[0][0][0], vbase_p, HDIM, CHUNK, EMB);
    tdm_load_2d(&MlB[0][0][0], mbase_p, CHUNK, 64, SEQ);
  }

  for (int kc = 0; kc < SEQ; kc += CHUNK) {
    const int buf = (kc >> 5) & 1;
    if (wave == 0) {
      if (kc + CHUNK < SEQ) {  // prefetch next chunk into the other buffer
        tdm_load_2d(&KlB[buf ^ 1][0][0], kbase_p + (size_t)(kc + CHUNK) * EMB, HDIM, CHUNK, EMB);
        tdm_load_2d(&VlB[buf ^ 1][0][0], vbase_p + (size_t)(kc + CHUNK) * EMB, HDIM, CHUNK, EMB);
        tdm_load_2d(&MlB[buf ^ 1][0][0], mbase_p + (size_t)(kc + CHUNK), CHUNK, 64, SEQ);
        __builtin_amdgcn_s_wait_tensorcnt(3);  // current chunk's 3 TDMs done
      } else {
        __builtin_amdgcn_s_wait_tensorcnt(0);
      }
    }
    __syncthreads();

    // S = Q K^T for two 16-key tiles (contraction over head_dim, 2 k-chunks).
    v8f s0 = {}, s1 = {};
    {
      v16bf b;
      LOAD_FRAG_CVT(b, &KlB[buf][lq][0], 0);        s0 = WMMA_BF16(aQ0, b, s0);
      LOAD_FRAG_CVT(b, &KlB[buf][lq][0], 32);       s0 = WMMA_BF16(aQ1, b, s0);
      LOAD_FRAG_CVT(b, &KlB[buf][16 + lq][0], 0);   s1 = WMMA_BF16(aQ0, b, s1);
      LOAD_FRAG_CVT(b, &KlB[buf][16 + lq][0], 32);  s1 = WMMA_BF16(aQ1, b, s1);
    }

    // Reference quirk: masked energies -> +1e-20 BEFORE the 1/32 scaling.
    float e0[8], e1[8];
#pragma unroll
    for (int r = 0; r < 8; ++r) {
      int qlocal = wave * 16 + r + 8 * half;
      e0[r] = MlB[buf][qlocal][lq]      ? s0[r] : 1e-20f;
      e1[r] = MlB[buf][qlocal][16 + lq] ? s1[r] : 1e-20f;
    }

    // Online softmax; rows live across the 16 lanes of each half-wave.
#pragma unroll
    for (int r = 0; r < 8; ++r) {
      float v = fmaxf(e0[r], e1[r]);
      v = fmaxf(v, __shfl_xor(v, 1, 32));
      v = fmaxf(v, __shfl_xor(v, 2, 32));
      v = fmaxf(v, __shfl_xor(v, 4, 32));
      v = fmaxf(v, __shfl_xor(v, 8, 32));
      float mnew  = fmaxf(mrow[r], v);
      float scale = __expf((mrow[r] - mnew) * inv_scale);
      mrow[r] = mnew;
      lrow[r] *= scale;
      acc[0][r] *= scale; acc[1][r] *= scale; acc[2][r] *= scale; acc[3][r] *= scale;
      float p0 = __expf((e0[r] - mnew) * inv_scale);
      float p1 = __expf((e1[r] - mnew) * inv_scale);
      Pl[wave][r + 8 * half][lq]      = (__bf16)p0;
      Pl[wave][r + 8 * half][16 + lq] = (__bf16)p1;
      float rs = p0 + p1;
      rs += __shfl_xor(rs, 1, 32);
      rs += __shfl_xor(rs, 2, 32);
      rs += __shfl_xor(rs, 4, 32);
      rs += __shfl_xor(rs, 8, 32);
      lrow[r] += rs;
    }

    // acc += P * V  (4 head-dim tiles).  V B-fragments: column d over the 32
    // keys -> strided fp32 LDS reads + cvt.  Same-wave DS ops stay in order.
    v16bf pa;
    LOAD_FRAG(pa, &Pl[wave][lq][0], 0);
#pragma unroll
    for (int t = 0; t < 4; ++t) {
      v16bf b;
#pragma unroll
      for (int j = 0; j < 8; ++j) {
        b[j]     = (__bf16)VlB[buf][kb + j][t * 16 + lq];
        b[j + 8] = (__bf16)VlB[buf][kb + 16 + j][t * 16 + lq];
      }
      acc[t] = WMMA_BF16(pa, b, acc[t]);
    }
    __syncthreads();  // all reads of buf done before its next TDM overwrite
  }

  // Normalize and write fp32 context to workspace.
#pragma unroll
  for (int r = 0; r < 8; ++r) {
    float invl = 1.0f / lrow[r];
    int qg = qbase + r + 8 * half;
    float* orow = X + ((size_t)n * SEQ + qg) * EMB + head_off;
    orow[lq]      = acc[0][r] * invl;
    orow[16 + lq] = acc[1][r] * invl;
    orow[32 + lq] = acc[2][r] * invl;
    orow[48 + lq] = acc[3][r] * invl;
  }
}

// ---------------------------------------------------------------------------
// Kernel 2: Out = X @ W^T + b.   grid = (N*SEQ/16, EMB/256), block = 128.
// W^T's column e is W's row e -> contiguous B fragments.  TDM double-buffers
// the A tile (16x32) and B tile (256x32) per k-chunk.
// ---------------------------------------------------------------------------
__global__ __launch_bounds__(128)
void proj_kernel(const float* __restrict__ X, const float* __restrict__ W,
                 const float* __restrict__ bias, float* __restrict__ Out) {
  __shared__ float Al[2][16][CHUNK];     //  4 KB
  __shared__ float Bl[2][256][CHUNK];    // 64 KB

  const int tid  = threadIdx.x;
  const int wave = tid >> 5;
  const int lane = tid & 31;
  const int lq   = lane & 15;
  const int half = lane >> 4;
  const int kb   = half * 8;

  const int mbase = blockIdx.x * 16;
  const int ebase = blockIdx.y * 256;

  const float* arow = X + (size_t)mbase * EMB;
  const float* brow = W + (size_t)ebase * EMB;

  v8f acc[4] = {{}, {}, {}, {}};

  if (wave == 0) {
    tdm_load_2d(&Al[0][0][0], arow, CHUNK, 16, EMB);
    tdm_load_2d(&Bl[0][0][0], brow, CHUNK, 256, EMB);
  }

  for (int kc = 0; kc < EMB; kc += CHUNK) {
    const int buf = (kc >> 5) & 1;
    if (wave == 0) {
      if (kc + CHUNK < EMB) {
        tdm_load_2d(&Al[buf ^ 1][0][0], arow + kc + CHUNK, CHUNK, 16, EMB);
        tdm_load_2d(&Bl[buf ^ 1][0][0], brow + kc + CHUNK, CHUNK, 256, EMB);
        __builtin_amdgcn_s_wait_tensorcnt(2);
      } else {
        __builtin_amdgcn_s_wait_tensorcnt(0);
      }
    }
    __syncthreads();

    v16bf a;
    LOAD_FRAG_CVT(a, &Al[buf][lq][0], 0);
#pragma unroll
    for (int t = 0; t < 4; ++t) {
      v16bf b;
      LOAD_FRAG_CVT(b, &Bl[buf][wave * 64 + t * 16 + lq][0], 0);
      acc[t] = WMMA_BF16(a, b, acc[t]);
    }
    __syncthreads();
  }

#pragma unroll
  for (int t = 0; t < 4; ++t) {
    int e = ebase + wave * 64 + t * 16 + lq;
    float bv = bias[e];
#pragma unroll
    for (int r = 0; r < 8; ++r) {
      int mr = mbase + r + 8 * half;
      Out[(size_t)mr * EMB + e] = acc[t][r] + bv;
    }
  }
}

// ---------------------------------------------------------------------------
extern "C" void kernel_launch(void* const* d_in, const int* in_sizes, int n_in,
                              void* d_out, int out_size, void* d_ws, size_t ws_size,
                              hipStream_t stream) {
  (void)in_sizes; (void)n_in; (void)out_size; (void)ws_size;
  const float* Vp = (const float*)d_in[0];   // values  (N, L, E) f32
  const float* Kp = (const float*)d_in[1];   // keys    (N, L, E) f32
  const float* Qp = (const float*)d_in[2];   // query   (N, L, E) f32
  const int*   Mp = (const int*)d_in[3];     // mask    (N, 1, L, L) i32
  const float* Wp = (const float*)d_in[4];   // W_out   (E, E) f32
  const float* bp = (const float*)d_in[5];   // b_out   (E,) f32
  // d_in[6] = heads (==16, baked in)

  float* X   = (float*)d_ws;    // context, N*L*E f32 (33.5 MB)
  float* Out = (float*)d_out;

  attn_fused_kernel<<<dim3(NBATCH * HEADS, SEQ / 64), 128, 0, stream>>>(Vp, Kp, Qp, Mp, X);
  proj_kernel<<<dim3((NBATCH * SEQ) / 16, EMB / 256), 128, 0, stream>>>(X, Wp, bp, Out);
}
